// Node_25546465476856
// MI455X (gfx1250) — compile-verified, matrix operand order checked by reading, not run on hardware
//
#include <hip/hip_runtime.h>
#include <hip/hip_bf16.h>
#include <math.h>

// ---------------------------------------------------------------------------
// MI455X (gfx1250, wave32) fused hierarchical-router forward.
//   Phase 1: h = relu(x@W1+b1) via v_wmma_f32_16x16x32_bf16, fused head
//            logits/softmax/route, builds per-expert compacted row lists.
//   Phase 2: per-expert gathered GEMM h_c = relu(h@Wc1[c]+bc1[c]) + head,
//            scatters results over out_h / out_logits for routed rows only.
// Precision: bf16 A/B fragments, f32 accumulate (memory-bound on HBM at
// 23.3 TB/s; f32 WMMA K=4 would be ~8x slower per MAC and compute-bound).
// ---------------------------------------------------------------------------

typedef __attribute__((ext_vector_type(16))) __bf16 v16bf;
typedef __attribute__((ext_vector_type(8)))  float  v8f;

#define KP 40          // LDS row stride in bf16 elems (80B, 16B-aligned, padded)
#define TILE_M 128

__device__ __forceinline__ unsigned short f2bf(float f) {
  unsigned int u = __float_as_uint(f);
  u += 0x7FFFu + ((u >> 16) & 1u);          // round-to-nearest-even
  return (unsigned short)(u >> 16);
}

// ISA 7.12.2 16-bit A/B fragment: lane holds row (A) / col (B) = lane&15,
// K halves: lanes 0-15 -> K[0..7],K[16..23]; lanes 16-31 -> K[8..15],K[24..31]
__device__ __forceinline__ v16bf load_frag(const unsigned short* row, int kb) {
  union { v16bf v; uint4 q[2]; } u;
  u.q[0] = *(const uint4*)(row + kb);       // ds_load_b128
  u.q[1] = *(const uint4*)(row + 16 + kb);  // ds_load_b128
  return u.v;
}

// ---------------------------------------------------------------------------
__global__ void __launch_bounds__(256)
init_counts(int* counts) {
  if (threadIdx.x < 8) counts[threadIdx.x] = 0;
}

// ---------------------------------------------------------------------------
__global__ void __launch_bounds__(256)
fused_root_kernel(const float* __restrict__ x,
                  const float* __restrict__ W1,
                  const float* __restrict__ b1,
                  const float* __restrict__ W2,
                  const float* __restrict__ b2,
                  const float* __restrict__ tau,
                  float* __restrict__ out_logits,
                  float* __restrict__ out_h,
                  int*   __restrict__ depths,
                  int*   __restrict__ counts,
                  int*   __restrict__ lists,
                  int Btot)
{
  __shared__ unsigned short xs[TILE_M * KP];   // x tile, bf16  (10 KB)
  __shared__ unsigned short wt[256 * KP];      // W1 K-slice, transposed (20 KB)
  __shared__ float W2s[256 * 8];               // head weights (8 KB)
  __shared__ float b1s[256];

  const int  t    = threadIdx.x;
  const int  lane = t & 31;
  const int  wave = t >> 5;
  const int  m0   = wave * 16;                 // wave's 16-row strip
  const int  colb = lane & 15;
  const int  half = lane >> 4;
  const int  kb   = half * 8;
  const long m0g  = (long)blockIdx.x * TILE_M;

  for (int i = t; i < 256 * 8; i += 256) W2s[i] = W2[i];
  b1s[t] = b1[t];

  v8f acc[16];
  const v8f vz = {0.f, 0.f, 0.f, 0.f, 0.f, 0.f, 0.f, 0.f};
#pragma unroll
  for (int i = 0; i < 16; ++i) acc[i] = vz;

  for (int k0 = 0; k0 < 256; k0 += 32) {
    // ---- stage x tile (128x32 f32 -> bf16), coalesced 64B/thread ----
    {
      const int m  = t >> 1;
      const int kk = (t & 1) * 16;
      const float* src = x + (m0g + m) * 256 + k0 + kk;
      float4 f0 = *(const float4*)(src + 0);
      float4 f1 = *(const float4*)(src + 4);
      float4 f2 = *(const float4*)(src + 8);
      float4 f3 = *(const float4*)(src + 12);
      unsigned short* d = &xs[m * KP + kk];
      d[0]=f2bf(f0.x); d[1]=f2bf(f0.y); d[2]=f2bf(f0.z); d[3]=f2bf(f0.w);
      d[4]=f2bf(f1.x); d[5]=f2bf(f1.y); d[6]=f2bf(f1.z); d[7]=f2bf(f1.w);
      d[8]=f2bf(f2.x); d[9]=f2bf(f2.y); d[10]=f2bf(f2.z); d[11]=f2bf(f2.w);
      d[12]=f2bf(f3.x); d[13]=f2bf(f3.y); d[14]=f2bf(f3.z); d[15]=f2bf(f3.w);
      if (k0 + 32 < 256)
        __builtin_prefetch(src + 32, 0, 1);   // global_prefetch_b8 next slice
    }
    // ---- stage W1[k0:k0+32, :] transposed -> wt[n][k] ----
#pragma unroll 4
    for (int i = 0; i < 32; ++i)
      wt[t * KP + i] = f2bf(W1[(k0 + i) * 256 + t]);
    __syncthreads();

    const v16bf a = load_frag(&xs[(m0 + colb) * KP], kb);
#pragma unroll
    for (int nt = 0; nt < 16; ++nt) {
      const v16bf b = load_frag(&wt[(nt * 16 + colb) * KP], kb);
      acc[nt] = __builtin_amdgcn_wmma_f32_16x16x32_bf16(
          false, a, false, b, (short)0, acc[nt], false, false);
    }
    __syncthreads();
  }

  // ---- epilogue: bias + relu, write h, accumulate head-logit partials ----
  float p[8][8];
#pragma unroll
  for (int r = 0; r < 8; ++r)
#pragma unroll
    for (int j = 0; j < 8; ++j) p[r][j] = 0.f;

#pragma unroll
  for (int nt = 0; nt < 16; ++nt) {
    const int col  = nt * 16 + colb;
    const float bs = b1s[col];
#pragma unroll
    for (int r = 0; r < 8; ++r) {
      float v = fmaxf(acc[nt][r] + bs, 0.f);
      const int rowl = m0 + r + 8 * half;           // C-layout: M = r + 8*half
      out_h[(m0g + rowl) * 256 + col] = v;
#pragma unroll
      for (int j = 0; j < 8; ++j) p[r][j] = fmaf(v, W2s[col * 8 + j], p[r][j]);
    }
  }
  // reduce across the 16 lanes sharing each row
#pragma unroll
  for (int off = 1; off < 16; off <<= 1)
#pragma unroll
    for (int r = 0; r < 8; ++r)
#pragma unroll
      for (int j = 0; j < 8; ++j)
        p[r][j] += __shfl_xor(p[r][j], off, 32);

  if (colb == 0) {                                  // lanes 0 and 16
    float b2r[8], taur[8];
#pragma unroll
    for (int j = 0; j < 8; ++j) { b2r[j] = b2[j]; taur[j] = tau[j]; }
#pragma unroll 1
    for (int r = 0; r < 8; ++r) {
      const long row = m0g + m0 + r + 8 * half;
      float lg[8], mx = -1e30f;
#pragma unroll
      for (int j = 0; j < 8; ++j) { lg[j] = p[r][j] + b2r[j]; mx = fmaxf(mx, lg[j]); }
      float e[8], s = 0.f;
#pragma unroll
      for (int j = 0; j < 8; ++j) { e[j] = __expf(lg[j] - mx); s += e[j]; }
      const float inv = 1.f / s;
      int best = -1; float bp = -1.f;
#pragma unroll
      for (int j = 0; j < 8; ++j) {
        const float pr = e[j] * inv;
        if (pr >= taur[j] && pr > bp) { bp = pr; best = j; }
      }
      float4* lo = (float4*)(out_logits + row * 8);
      lo[0] = make_float4(lg[0], lg[1], lg[2], lg[3]);
      lo[1] = make_float4(lg[4], lg[5], lg[6], lg[7]);
      depths[row] = (best >= 0) ? 1 : 0;
      if (best >= 0) {
        const int pos = atomicAdd(&counts[best], 1);
        lists[(size_t)best * Btot + pos] = (int)row;
      }
    }
  }
}

// ---------------------------------------------------------------------------
__global__ void __launch_bounds__(256)
expert_kernel(const float* __restrict__ Wc1,
              const float* __restrict__ bc1,
              const float* __restrict__ Wc2,
              const float* __restrict__ bc2,
              float* __restrict__ out_logits,
              float* __restrict__ out_h,       // read h (phase 1), scatter h_c
              const int* __restrict__ counts,
              const int* __restrict__ lists,
              int Btot, int tilesPerExpert)
{
  const int expert = blockIdx.x / tilesPerExpert;
  const int tile   = blockIdx.x % tilesPerExpert;
  const int count  = counts[expert];
  const int base   = tile * TILE_M;
  if (base >= count) return;                        // empty tile: 1 load, exit
  const int nrows  = min(TILE_M, count - base);

  __shared__ int sidx[TILE_M];
  __shared__ unsigned short xs[TILE_M * KP];
  __shared__ unsigned short wt[256 * KP];
  __shared__ float W2s[256 * 8];
  __shared__ float b1s[256];

  const int t    = threadIdx.x;
  const int lane = t & 31;
  const int wave = t >> 5;
  const int m0   = wave * 16;
  const int colb = lane & 15;
  const int half = lane >> 4;
  const int kb   = half * 8;

  const float* W1e = Wc1 + (size_t)expert * 256 * 256;
  const float* W2e = Wc2 + (size_t)expert * 256 * 8;

  if (t < TILE_M)
    sidx[t] = lists[(size_t)expert * Btot + base + min(t, nrows - 1)];
  for (int i = t; i < 256 * 8; i += 256) W2s[i] = W2e[i];
  b1s[t] = bc1[expert * 256 + t];
  __syncthreads();

  v8f acc[16];
  const v8f vz = {0.f, 0.f, 0.f, 0.f, 0.f, 0.f, 0.f, 0.f};
#pragma unroll
  for (int i = 0; i < 16; ++i) acc[i] = vz;

  for (int k0 = 0; k0 < 256; k0 += 32) {
    {
      const int m  = t >> 1;
      const int kk = (t & 1) * 16;
      const float* src = out_h + (size_t)sidx[m] * 256 + k0 + kk;  // gather
      float4 f0 = *(const float4*)(src + 0);
      float4 f1 = *(const float4*)(src + 4);
      float4 f2 = *(const float4*)(src + 8);
      float4 f3 = *(const float4*)(src + 12);
      unsigned short* d = &xs[m * KP + kk];
      d[0]=f2bf(f0.x); d[1]=f2bf(f0.y); d[2]=f2bf(f0.z); d[3]=f2bf(f0.w);
      d[4]=f2bf(f1.x); d[5]=f2bf(f1.y); d[6]=f2bf(f1.z); d[7]=f2bf(f1.w);
      d[8]=f2bf(f2.x); d[9]=f2bf(f2.y); d[10]=f2bf(f2.z); d[11]=f2bf(f2.w);
      d[12]=f2bf(f3.x); d[13]=f2bf(f3.y); d[14]=f2bf(f3.z); d[15]=f2bf(f3.w);
    }
#pragma unroll 4
    for (int i = 0; i < 32; ++i)
      wt[t * KP + i] = f2bf(W1e[(k0 + i) * 256 + t]);
    __syncthreads();

    const v16bf a = load_frag(&xs[(m0 + colb) * KP], kb);
#pragma unroll
    for (int nt = 0; nt < 16; ++nt) {
      const v16bf b = load_frag(&wt[(nt * 16 + colb) * KP], kb);
      acc[nt] = __builtin_amdgcn_wmma_f32_16x16x32_bf16(
          false, a, false, b, (short)0, acc[nt], false, false);
    }
    __syncthreads();
  }

  float p[8][8];
#pragma unroll
  for (int r = 0; r < 8; ++r)
#pragma unroll
    for (int j = 0; j < 8; ++j) p[r][j] = 0.f;

#pragma unroll
  for (int nt = 0; nt < 16; ++nt) {
    const int col  = nt * 16 + colb;
    const float bs = b1s[col];
#pragma unroll
    for (int r = 0; r < 8; ++r) {
      float v = fmaxf(acc[nt][r] + bs, 0.f);
      const int mloc = m0 + r + 8 * half;
      if (mloc < nrows)
        out_h[(size_t)sidx[mloc] * 256 + col] = v;   // scatter h_c
#pragma unroll
      for (int j = 0; j < 8; ++j) p[r][j] = fmaf(v, W2s[col * 8 + j], p[r][j]);
    }
  }
#pragma unroll
  for (int off = 1; off < 16; off <<= 1)
#pragma unroll
    for (int r = 0; r < 8; ++r)
#pragma unroll
      for (int j = 0; j < 8; ++j)
        p[r][j] += __shfl_xor(p[r][j], off, 32);

  if (colb == 0) {
    float b2r[8];
#pragma unroll
    for (int j = 0; j < 8; ++j) b2r[j] = bc2[expert * 8 + j];
#pragma unroll 1
    for (int r = 0; r < 8; ++r) {
      const int mloc = m0 + r + 8 * half;
      if (mloc < nrows) {
        const long row = sidx[mloc];
        float4* lo = (float4*)(out_logits + row * 8);
        lo[0] = make_float4(p[r][0]+b2r[0], p[r][1]+b2r[1], p[r][2]+b2r[2], p[r][3]+b2r[3]);
        lo[1] = make_float4(p[r][4]+b2r[4], p[r][5]+b2r[5], p[r][6]+b2r[6], p[r][7]+b2r[7]);
      }
    }
  }
}

// ---------------------------------------------------------------------------
extern "C" void kernel_launch(void* const* d_in, const int* in_sizes, int n_in,
                              void* d_out, int out_size, void* d_ws, size_t ws_size,
                              hipStream_t stream)
{
  const float* x   = (const float*)d_in[0];
  const float* W1  = (const float*)d_in[1];
  const float* b1  = (const float*)d_in[2];
  const float* W2  = (const float*)d_in[3];
  const float* b2  = (const float*)d_in[4];
  const float* Wc1 = (const float*)d_in[5];
  const float* bc1 = (const float*)d_in[6];
  const float* Wc2 = (const float*)d_in[7];
  const float* bc2 = (const float*)d_in[8];
  const float* tau = (const float*)d_in[9];

  const int Btot = in_sizes[0] / 256;            // B = 131072

  float* out_logits = (float*)d_out;                       // [B,8]
  float* out_h      = out_logits + (size_t)Btot * 8;       // [B,256]
  int*   depths     = (int*)(out_h + (size_t)Btot * 256);  // [B] int32 bits

  int* counts = (int*)d_ws;          // 8 counters (256B-padded)
  int* lists  = counts + 64;         // [C, B] routed-row indices (~4.2 MB)

  hipLaunchKernelGGL(init_counts, dim3(1), dim3(256), 0, stream, counts);

  hipLaunchKernelGGL(fused_root_kernel, dim3(Btot / TILE_M), dim3(256), 0, stream,
                     x, W1, b1, W2, b2, tau,
                     out_logits, out_h, depths, counts, lists, Btot);

  const int tilesPerExpert = (Btot + TILE_M - 1) / TILE_M;
  hipLaunchKernelGGL(expert_kernel, dim3(8 * tilesPerExpert), dim3(256), 0, stream,
                     Wc1, bc1, Wc2, bc2,
                     out_logits, out_h, counts, lists, Btot, tilesPerExpert);
}
